// InteractionNet_2302102471382
// MI455X (gfx1250) — matrix operand order repaired and checked by reading, not compiled, not run on hardware
//
#include <hip/hip_runtime.h>

// Fused 6-layer GraphConv (A@X@W + b) + final 128->3 projection + residual.
// One workgroup = 16 batch items resident in LDS (f16) across all layers.
// GEMMs on v_wmma_f32_16x16x32_f16 (f32 accumulate). Weights are preprocessed
// once (f32 -> f16, transposed/padded) into d_ws by a prep kernel; the main
// kernel double-buffers W in LDS and prefetches layer l+1's weights with
// global_load_async_to_lds_b128 while layer l computes.

#define BATCH   16384
#define NN      29
#define DM      128
#define BT      16          // batch items per workgroup
#define NPAD    32          // nodes padded to 32
#define THREADS 256         // 8 waves (wave32)

typedef __attribute__((ext_vector_type(16))) _Float16 v16h;
typedef __attribute__((ext_vector_type(8)))  _Float16 v8h;
typedef __attribute__((ext_vector_type(8)))  float    v8f;

// LDS layout (halves):
//   X   [BT*NPAD][DM]   row-major activations
//   Zt  [BT][DM][NPAD]  transposed X@W result (feature-major, node inner)
//   Wb0/Wb1 [DM][DM]    double-buffered W^T (f16, pre-transposed in d_ws)
#define X_ELEMS  (BT * NPAD * DM)
#define ZT_ELEMS (BT * DM * NPAD)
#define WL_ELEMS (DM * DM)
#define AP_ELEMS (NPAD * NPAD)
#define SMEM_BYTES ((X_ELEMS + ZT_ELEMS + 2 * WL_ELEMS) * 2)   // 327680 = 320KB

// ---------------- CDNA5 async global->LDS copy ----------------
__device__ __forceinline__ unsigned lds_off(const void* p) {
    return (unsigned)(size_t)p;            // generic LDS ptr: addr[31:0] = LDS offset
}
__device__ __forceinline__ void async_ld_b128(unsigned ldsaddr, const void* gaddr) {
    asm volatile("global_load_async_to_lds_b128 %0, %1, off"
                 :: "v"(ldsaddr), "v"(gaddr) : "memory");
}
__device__ __forceinline__ void wait_async0() {
    asm volatile("s_wait_asynccnt 0" ::: "memory");
}

// A-operand fragment (16xK f16): p = row base + kb + lanehalf*16.
__device__ __forceinline__ v16h load_frag_a(const _Float16* p) {
    v8h lo = *(const v8h*)(p);
    v8h hi = *(const v8h*)(p + 8);
    v16h r;
#pragma unroll
    for (int i = 0; i < 8; ++i) { r[i] = lo[i]; r[i + 8] = hi[i]; }
    return r;
}
// B-operand fragment: 16 contiguous halves of one column image (32B aligned).
__device__ __forceinline__ v16h load_frag_b(const _Float16* p) {
    return *(const v16h*)p;
}

// ---------------- prep: W -> f16 W^T (zero-padded K), A -> f16 32x32 ----------------
extern "C" __global__ void gcn6_prep(
    const float* W1, const float* W2, const float* W3,
    const float* W4, const float* W5, const float* W6,
    const float* A1, const float* A2, const float* A3,
    const float* A4, const float* A5, const float* A6,
    _Float16* wsW, _Float16* wsA)
{
    const float* Ws[6] = {W1, W2, W3, W4, W5, W6};
    const float* As[6] = {A1, A2, A3, A4, A5, A6};
    int idx = blockIdx.x * blockDim.x + threadIdx.x;
    if (idx < 6 * WL_ELEMS) {
        int l = idx >> 14;            // / WL_ELEMS
        int r = idx & (WL_ELEMS - 1);
        int e = r >> 7, d = r & 127;
        int Kin = l ? DM : 3;
        float v = (d < Kin) ? Ws[l][d * DM + e] : 0.0f;
        wsW[idx] = (_Float16)v;       // wsW[l][e][d] = W_l[d][e]
    } else if (idx < 6 * WL_ELEMS + 6 * AP_ELEMS) {
        int j = idx - 6 * WL_ELEMS;
        int l = j >> 10;              // / AP_ELEMS
        int r = j & (AP_ELEMS - 1);
        int n = r >> 5, m = r & 31;
        float v = (n < NN && m < NN) ? As[l][n * NN + m] : 0.0f;
        wsA[j] = (_Float16)v;         // wsA[l][n][m], zero-padded
    }
}

// ---------------- main fused kernel ----------------
extern "C" __global__ void __launch_bounds__(THREADS)
gcn6_fused(const float* __restrict__ pose,
           const float* __restrict__ b1, const float* __restrict__ b2,
           const float* __restrict__ b3, const float* __restrict__ b4,
           const float* __restrict__ b5, const float* __restrict__ b6,
           const float* __restrict__ Wtp, const float* __restrict__ btp,
           const _Float16* __restrict__ wsW, const _Float16* __restrict__ wsA,
           float* __restrict__ out)
{
    extern __shared__ _Float16 smem[];
    _Float16* X   = smem;
    _Float16* Zt  = X + X_ELEMS;
    _Float16* Wb0 = Zt + ZT_ELEMS;
    _Float16* Wb1 = Wb0 + WL_ELEMS;

    const int tid  = threadIdx.x;
    const int wave = tid >> 5;
    const int lane = tid & 31;
    const int hf   = lane >> 4;     // lane half (0: lanes 0-15, 1: lanes 16-31)
    const int l16  = lane & 15;
    const int b0   = blockIdx.x * BT;

    // ---- kick off async copy of W_0^T (f16, 32KB) into Wb0 ----
    for (int i = tid; i < WL_ELEMS / 8; i += THREADS)
        async_ld_b128(lds_off(Wb0 + i * 8), wsW + i * 8);

    // ---- stage layer-1 input: X[row][0..31] = padded init_pose (K 3->32) ----
    for (int idx = tid; idx < BT * NPAD * 32; idx += THREADS) {
        int k    = idx & 31;
        int row  = idx >> 5;
        int node = row & (NPAD - 1);
        int b    = row >> 5;
        float v = 0.0f;
        if (node < NN && k < 3) v = pose[((b0 + b) * NN + node) * 3 + k];
        X[row * DM + k] = (_Float16)v;
    }
    wait_async0();
    __syncthreads();

    const float* bs[6] = {b1, b2, b3, b4, b5, b6};

#pragma unroll
    for (int layer = 0; layer < 6; ++layer) {
        _Float16* Wcur = (layer & 1) ? Wb1 : Wb0;
        _Float16* Wnxt = (layer & 1) ? Wb0 : Wb1;
        const int ksteps = (layer == 0) ? 1 : 4;

        // ---- async prefetch next layer's W^T into the other buffer ----
        if (layer < 5) {
            const _Float16* src = wsW + (layer + 1) * WL_ELEMS;
            for (int i = tid; i < WL_ELEMS / 8; i += THREADS)
                async_ld_b128(lds_off(Wnxt + i * 8), src + i * 8);
        }

        // ---- Z^T = (X @ W)^T : each wave owns 16 output features ----
        v16h wb[4];
#pragma unroll
        for (int s = 0; s < 4; ++s)
            if (s < ksteps)
                wb[s] = load_frag_b(&Wcur[(wave * 16 + l16) * DM + s * 32 + hf * 16]);

        for (int mt = 0; mt < 2 * BT; ++mt) {
            const int row = mt * 16 + l16;          // padded (batch,node) row
            v8f acc = {};
#pragma unroll
            for (int s = 0; s < 4; ++s) {
                if (s < ksteps) {
                    v16h xa = load_frag_a(&X[row * DM + s * 32 + hf * 16]);
                    acc = __builtin_amdgcn_wmma_f32_16x16x32_f16(
                        false, xa, false, wb[s], (short)0, acc, false, false);
                }
            }
            // D rows = 8 consecutive node rows per lane -> contiguous in Zt
            const int b     = mt >> 1;
            const int e     = wave * 16 + l16;
            const int mbase = (mt & 1) * 16 + hf * 8;
            v8h h;
#pragma unroll
            for (int r = 0; r < 8; ++r) h[r] = (_Float16)acc[r];
            *(v8h*)&Zt[(b * DM + e) * NPAD + mbase] = h;
        }
        __syncthreads();

        // ---- X = A @ Z + bias, computed as (Z^T @ A^T); wave owns 16 d-rows ----
        const _Float16* Ag = wsA + layer * AP_ELEMS;      // padded A, row-major
        v16h abf0 = *(const v16h*)(Ag + l16 * NPAD + hf * 16);        // n-tile 0
        v16h abf1 = *(const v16h*)(Ag + (16 + l16) * NPAD + hf * 16); // n-tile 1
        const int dt = wave;
        const v8f bbv = *(const v8f*)(bs[layer] + dt * 16 + hf * 8);

        for (int b = 0; b < BT; ++b) {
            const int drow = dt * 16 + l16;
            v16h za = load_frag_a(&Zt[(b * DM + drow) * NPAD + hf * 16]);
#pragma unroll
            for (int nt = 0; nt < 2; ++nt) {
                v8f acc = {};
                acc = __builtin_amdgcn_wmma_f32_16x16x32_f16(
                    false, za, false, (nt ? abf1 : abf0), (short)0, acc, false, false);
                // D rows = 8 consecutive d per lane -> contiguous in row-major X
                const int n = nt * 16 + l16;
                v8h h;
#pragma unroll
                for (int r = 0; r < 8; ++r) h[r] = (_Float16)(acc[r] + bbv[r]);
                *(v8h*)&X[(b * NPAD + n) * DM + dt * 16 + hf * 8] = h;
            }
        }
        wait_async0();          // prefetched W for layer+1 is now resident
        __syncthreads();
    }

    // ---- out = pose + x @ Wt + bt  (tiny 128->3 tail, VALU f32) ----
    for (int idx = tid; idx < BT * NN * 3; idx += THREADS) {
        int c = idx % 3;
        int t = idx / 3;
        int n = t % NN;
        int b = t / NN;
        float acc = btp[c];
        const _Float16* xr = &X[(b * NPAD + n) * DM];
#pragma unroll 8
        for (int d = 0; d < DM; ++d) acc += (float)xr[d] * Wtp[d * 3 + c];
        int g = ((b0 + b) * NN + n) * 3 + c;
        out[g] = pose[g] + acc;
    }
}

extern "C" void kernel_launch(void* const* d_in, const int* in_sizes, int n_in,
                              void* d_out, int out_size, void* d_ws, size_t ws_size,
                              hipStream_t stream) {
    (void)in_sizes; (void)n_in; (void)out_size; (void)ws_size;

    const float* pose = (const float*)d_in[0];
    const float* A[6];
    for (int i = 0; i < 6; ++i) A[i] = (const float*)d_in[1 + i];
    const float* W[6];
    const float* bv[6];
    for (int i = 0; i < 6; ++i) {
        W[i]  = (const float*)d_in[7 + 2 * i];
        bv[i] = (const float*)d_in[8 + 2 * i];
    }
    const float* Wt = (const float*)d_in[19];
    const float* bt = (const float*)d_in[20];
    float* out = (float*)d_out;

    _Float16* wsW = (_Float16*)d_ws;                 // 6*128*128 f16 = 192KB
    _Float16* wsA = wsW + 6 * WL_ELEMS;              // 6*32*32  f16 = 12KB

    // preprocess weights/adjacency (runs every call; deterministic)
    {
        int total = 6 * WL_ELEMS + 6 * AP_ELEMS;
        dim3 grid((total + THREADS - 1) / THREADS), block(THREADS);
        hipLaunchKernelGGL(gcn6_prep, grid, block, 0, stream,
                           W[0], W[1], W[2], W[3], W[4], W[5],
                           A[0], A[1], A[2], A[3], A[4], A[5],
                           wsW, wsA);
    }

    (void)hipFuncSetAttribute((const void*)gcn6_fused,
                              hipFuncAttributeMaxDynamicSharedMemorySize,
                              SMEM_BYTES);

    dim3 grid(BATCH / BT), block(THREADS);
    hipLaunchKernelGGL(gcn6_fused, grid, block, SMEM_BYTES, stream,
                       pose,
                       bv[0], bv[1], bv[2], bv[3], bv[4], bv[5],
                       Wt, bt, wsW, wsA, out);
}
